// QuantizedLinear_57741540327706
// MI455X (gfx1250) — compile-verified, matrix operand order checked by reading, not run on hardware
//
#include <hip/hip_runtime.h>
#include <hip/hip_bf16.h>
#include <stdint.h>

// ---------------------------------------------------------------------------
// NF4 quantized linear: out[8192,4096] = bf16( x[8192,4096] @ w[4096,4096]^T + bias )
//   Pass 1: dequant NF4 codes -> bf16 W in workspace (32 MB)
//   Pass 2: cast x fp32 -> bf16 in workspace (64 MB)
//   Pass 3: WMMA bf16 GEMM, 128x128 block tile, BK=32,
//           TDM (tensor_load_to_lds) double-buffered LDS pipeline,
//           single issuing wave behind a *scalar* branch (TDM ignores EXEC)
// ---------------------------------------------------------------------------

typedef __attribute__((ext_vector_type(16))) __bf16 v16bf;
typedef __attribute__((ext_vector_type(8)))  __bf16 v8bf;
typedef __attribute__((ext_vector_type(8)))  float  v8f;
typedef __attribute__((ext_vector_type(4)))  unsigned int v4u;
typedef __attribute__((ext_vector_type(8)))  unsigned int v8u;

#define IN_F  4096
#define OUT_F 4096
#define MROWS 8192   // 4 * 2048
#define BM 128
#define BN 128
#define BK 32

__device__ __constant__ float NF4_TBL[16] = {
    -1.0f, -0.6961928009986877f, -0.5250730514526367f, -0.39491748809814453f,
    -0.28444138169288635f, -0.18477343022823334f, -0.09105003625154495f, 0.0f,
    0.07958029955625534f, 0.16093020141124725f, 0.24611230194568634f,
    0.33791524171829224f, 0.44070982933044434f, 0.5626170039176941f,
    0.7229568362236023f, 1.0f};

__device__ __forceinline__ unsigned short f2bf(float f) {
  union { float f; uint32_t u; } v; v.f = f;
  uint32_t u = v.u;
  u += 0x7FFFu + ((u >> 16) & 1u);   // round-to-nearest-even
  return (unsigned short)(u >> 16);
}

// ---- Pass 1: NF4 double-dequant -> bf16 weights ---------------------------
__global__ __launch_bounds__(256) void nf4_dequant_kernel(
    const int* __restrict__ codes, const int* __restrict__ amax_q,
    const float* __restrict__ scale, unsigned short* __restrict__ wout) {
  size_t t  = (size_t)blockIdx.x * blockDim.x + threadIdx.x;
  size_t e0 = t * 8;
  float am = (float)amax_q[e0 >> 6] * (scale[0] * (1.0f / 127.0f));
  const int4* cp = (const int4*)(codes + e0);
  int4 ca = cp[0];
  int4 cb = cp[1];
  union { unsigned short s[8]; uint4 q; } o;
  o.s[0] = f2bf(NF4_TBL[ca.x & 15] * am);
  o.s[1] = f2bf(NF4_TBL[ca.y & 15] * am);
  o.s[2] = f2bf(NF4_TBL[ca.z & 15] * am);
  o.s[3] = f2bf(NF4_TBL[ca.w & 15] * am);
  o.s[4] = f2bf(NF4_TBL[cb.x & 15] * am);
  o.s[5] = f2bf(NF4_TBL[cb.y & 15] * am);
  o.s[6] = f2bf(NF4_TBL[cb.z & 15] * am);
  o.s[7] = f2bf(NF4_TBL[cb.w & 15] * am);
  *(uint4*)(wout + e0) = o.q;
}

// ---- Pass 2: x fp32 -> bf16 -----------------------------------------------
__global__ __launch_bounds__(256) void f32_to_bf16_kernel(
    const float* __restrict__ x, unsigned short* __restrict__ xo) {
  size_t t  = (size_t)blockIdx.x * blockDim.x + threadIdx.x;
  size_t e0 = t * 8;
  const float4* fp = (const float4*)(x + e0);
  float4 a = fp[0];
  float4 b = fp[1];
  union { unsigned short s[8]; uint4 q; } o;
  o.s[0] = f2bf(a.x); o.s[1] = f2bf(a.y); o.s[2] = f2bf(a.z); o.s[3] = f2bf(a.w);
  o.s[4] = f2bf(b.x); o.s[5] = f2bf(b.y); o.s[6] = f2bf(b.z); o.s[7] = f2bf(b.w);
  *(uint4*)(xo + e0) = o.q;
}

// ---- TDM: issue a 2D bf16 tile load into LDS (per-wave instruction) -------
// D# per CDNA5 ISA (2 groups => up to 2D tensor):
//  group0: [1:0]=count=1 | [63:32]=lds_addr | [120:64]=global_addr | [127:126]=type=2
//  group1: [17:16]=data_size(1=2B) | tensor_dim0 | tensor_dim1 |
//          tile_dim0 | tile_dim1 | tensor_dim0_stride
__device__ __forceinline__ void tdm_load_tile_2d(
    unsigned int lds_addr, const void* gtile,
    unsigned int tensor_d0, unsigned int tensor_d1, unsigned int stride_d0,
    unsigned int tile_d0, unsigned int tile_d1) {
  unsigned long long ga = (unsigned long long)(uintptr_t)gtile;
  v4u g0;
  g0[0] = 0x1u;                                   // count=1, user descriptor
  g0[1] = lds_addr;                               // LDS byte address
  g0[2] = (unsigned int)ga;                       // global_addr[31:0]
  g0[3] = (unsigned int)(ga >> 32) | (2u << 30);  // global_addr[56:32] | type=2
  v8u g1;
  g1[0] = (1u << 16);                             // wg_mask=0, data_size=2B
  g1[1] = (tensor_d0 & 0xFFFFu) << 16;            // tensor_dim0[15:0]
  g1[2] = (tensor_d0 >> 16) | ((tensor_d1 & 0xFFFFu) << 16);
  g1[3] = (tensor_d1 >> 16) | ((tile_d0 & 0xFFFFu) << 16);   // tile_dim0
  g1[4] = tile_d1 & 0xFFFFu;                      // tile_dim1 (tile_dim2=0)
  g1[5] = stride_d0;                              // tensor_dim0_stride[31:0]
  g1[6] = 0u;
  g1[7] = 0u;
  asm volatile("tensor_load_to_lds %0, %1" :: "s"(g0), "s"(g1) : "memory");
}

// ---- Pass 3: WMMA bf16 GEMM with TDM-fed LDS double buffering -------------
__global__ __launch_bounds__(256) void nf4_wmma_gemm_kernel(
    const unsigned short* __restrict__ X,   // [MROWS, IN_F] bf16
    const unsigned short* __restrict__ W,   // [OUT_F, IN_F] bf16
    const float* __restrict__ bias,         // [OUT_F] f32
    unsigned short* __restrict__ out) {     // [MROWS, OUT_F] bf16
  __shared__ __align__(16) unsigned short sA[2][BM * BK];
  __shared__ __align__(16) unsigned short sB[2][BN * BK];

  const int tid   = threadIdx.x;
  const int lane  = tid & 31;
  const int wave  = tid >> 5;
  const int waveM = wave >> 2;   // 0..1 -> 64 M-rows each
  const int waveN = wave & 3;    // 0..3 -> 32 N-cols each
  const int half  = lane >> 4;   // K-half selector per WMMA 16-bit layout
  const int l16   = lane & 15;

  // Wave-uniform SGPR predicate -> guarantees a *scalar branch* around the
  // TDM asm (TDM ignores EXEC, so predication would duplicate the DMA 8x).
  const bool issuer = (__builtin_amdgcn_readfirstlane(wave) == 0);

  const int m0 = blockIdx.y * BM;
  const int n0 = blockIdx.x * BN;

  const unsigned int ldsA[2] = {(unsigned int)(uintptr_t)&sA[0][0],
                                (unsigned int)(uintptr_t)&sA[1][0]};
  const unsigned int ldsB[2] = {(unsigned int)(uintptr_t)&sB[0][0],
                                (unsigned int)(uintptr_t)&sB[1][0]};
  const unsigned short* gA = X + (size_t)m0 * IN_F;   // tile row origin
  const unsigned short* gB = W + (size_t)n0 * IN_F;

  // prologue: issuing wave DMAs tile kt=0 into buffer 0, fences, barrier
  if (issuer) {
    tdm_load_tile_2d(ldsA[0], gA, IN_F, MROWS, IN_F, BK, BM);
    tdm_load_tile_2d(ldsB[0], gB, IN_F, OUT_F, IN_F, BK, BN);
    __builtin_amdgcn_s_wait_tensorcnt(0);
  }
  __syncthreads();

  v8f acc[4][2];
  {
    v8f z = {0.f, 0.f, 0.f, 0.f, 0.f, 0.f, 0.f, 0.f};
#pragma unroll
    for (int mi = 0; mi < 4; ++mi)
#pragma unroll
      for (int ni = 0; ni < 2; ++ni) acc[mi][ni] = z;
  }

  const int NKT = IN_F / BK;   // 128 K-steps
  int buf = 0;
#pragma unroll 1   // body has a barrier; unrolling only causes acc reg churn
  for (int kt = 0; kt < NKT; ++kt) {
    // kick off DMA of the next K-tile into the other buffer while computing
    if (issuer && kt + 1 < NKT) {
      const int nb = buf ^ 1;
      tdm_load_tile_2d(ldsA[nb], gA + (size_t)(kt + 1) * BK,
                       IN_F, MROWS, IN_F, BK, BM);
      tdm_load_tile_2d(ldsB[nb], gB + (size_t)(kt + 1) * BK,
                       IN_F, OUT_F, IN_F, BK, BN);
    }

    // LDS -> fragments per 16-bit WMMA layout: lane holds row l16, two 16B
    // chunks at K = half*8 and K = 16 + half*8.
    const unsigned short* sAb = &sA[buf][0];
    const unsigned short* sBb = &sB[buf][0];
    union { v16bf v; v8bf h[2]; } afrag[4], bfrag[2];
#pragma unroll
    for (int mi = 0; mi < 4; ++mi) {
      const unsigned short* p =
          sAb + (waveM * 64 + mi * 16 + l16) * BK + half * 8;
      afrag[mi].h[0] = *(const v8bf*)(p);
      afrag[mi].h[1] = *(const v8bf*)(p + 16);
    }
#pragma unroll
    for (int ni = 0; ni < 2; ++ni) {
      const unsigned short* p =
          sBb + (waveN * 32 + ni * 16 + l16) * BK + half * 8;
      bfrag[ni].h[0] = *(const v8bf*)(p);
      bfrag[ni].h[1] = *(const v8bf*)(p + 16);
    }

#pragma unroll
    for (int mi = 0; mi < 4; ++mi)
#pragma unroll
      for (int ni = 0; ni < 2; ++ni)
        acc[mi][ni] = __builtin_amdgcn_wmma_f32_16x16x32_bf16(
            false, afrag[mi].v, false, bfrag[ni].v,
            (short)0, acc[mi][ni], false, false);

    // issuing wave fences the DMA; barrier publishes the new buffer
    if (issuer && kt + 1 < NKT) {
      __builtin_amdgcn_s_wait_tensorcnt(0);
    }
    __syncthreads();
    buf ^= 1;
  }

  // Epilogue: C/D layout = VGPR v holds M = v + 8*half, N = l16; add bias, bf16
#pragma unroll
  for (int ni = 0; ni < 2; ++ni) {
    const int n = n0 + waveN * 32 + ni * 16 + l16;
    const float bv = bias[n];
#pragma unroll
    for (int mi = 0; mi < 4; ++mi) {
      const int mb = m0 + waveM * 64 + mi * 16 + half * 8;
#pragma unroll
      for (int v = 0; v < 8; ++v) {
        out[(size_t)(mb + v) * OUT_F + n] = f2bf(acc[mi][ni][v] + bv);
      }
    }
  }
}

// ---------------------------------------------------------------------------
extern "C" void kernel_launch(void* const* d_in, const int* in_sizes, int n_in,
                              void* d_out, int out_size, void* d_ws, size_t ws_size,
                              hipStream_t stream) {
  const float* x      = (const float*)d_in[0];   // [4,2048,4096] f32
  const int*   wq     = (const int*)d_in[1];     // [4096,4096] int32 codes
  const int*   amax   = (const int*)d_in[2];     // [262144] int32
  const float* ascale = (const float*)d_in[3];   // [1] f32
  const float* bias   = (const float*)d_in[4];   // [4096] f32
  unsigned short* out = (unsigned short*)d_out;  // [4,2048,4096] bf16

  unsigned short* wsW = (unsigned short*)d_ws;               // 32 MB bf16 W
  unsigned short* wsX = wsW + (size_t)OUT_F * IN_F;          // 64 MB bf16 X

  const int deqBlocks = (OUT_F / 256) * (IN_F / 8);          // 8192
  nf4_dequant_kernel<<<deqBlocks, 256, 0, stream>>>(wq, amax, ascale, wsW);

  const int castBlocks = (MROWS / 256) * (IN_F / 8);         // 16384
  f32_to_bf16_kernel<<<castBlocks, 256, 0, stream>>>(x, wsX);

  dim3 grid(OUT_F / BN, MROWS / BM);                         // (32, 64)
  nf4_wmma_gemm_kernel<<<grid, 256, 0, stream>>>(wsX, wsW, bias, out);
}
